// SpatialSelection_37306085933611
// MI455X (gfx1250) — compile-verified, matrix operand order checked by reading, not run on hardware
//
#include <hip/hip_runtime.h>
#include <math.h>

typedef __attribute__((ext_vector_type(16))) _Float16 v16h;
typedef __attribute__((ext_vector_type(8)))  _Float16 v8h;
typedef __attribute__((ext_vector_type(8)))  float    v8f;

#define C_DIM    512
#define N_PATCH  196
#define N_PAD    208          // 13 * 16
#define N_HEADS  4
#define HEAD_DIM 128
#define TOPM     10
#define B_DIM    16
#define T_DIM    60
#define SCALE    0.08838834764831845f   // 1/sqrt(128)

#define A_ELEMS  (N_PAD * C_DIM)        // 106496 f16
#define B_ELEMS  (16 * C_DIM)           // 8192 f16

// ---------------------------------------------------------------------------
// Phase 1: q[e] = dot(qst_feat[0,:], Wq[e,:]) + bq[e]        (512 outputs)
// ---------------------------------------------------------------------------
__global__ void k_qproj(const float* __restrict__ qst, const float* __restrict__ Wq,
                        const float* __restrict__ bq, float* __restrict__ qout) {
  int e = blockIdx.x * blockDim.x + threadIdx.x;
  if (e >= C_DIM) return;
  const float* wrow = Wq + (size_t)e * C_DIM;
  float s = bq[e];
  for (int c = 0; c < C_DIM; ++c) s = fmaf(qst[c], wrow[c], s);
  qout[e] = s;
}

// ---------------------------------------------------------------------------
// Phase 2: u[h][c] = sum_{i<128} q[h*128+i] * Wk[h*128+i][c]  (per-head folded
// projection vector), bias[h] = sum_{i<128} q[h*128+i] * bk[h*128+i]
// ---------------------------------------------------------------------------
__global__ void k_uproj(const float* __restrict__ q, const float* __restrict__ Wk,
                        const float* __restrict__ bk, float* __restrict__ u,
                        float* __restrict__ bias) {
  int tid = blockIdx.x * blockDim.x + threadIdx.x;
  if (tid >= N_HEADS * C_DIM) return;
  int h = tid >> 9;
  int c = tid & (C_DIM - 1);
  float s = 0.f;
  for (int i = 0; i < HEAD_DIM; ++i)
    s = fmaf(q[h * HEAD_DIM + i], Wk[(size_t)(h * HEAD_DIM + i) * C_DIM + c], s);
  u[tid] = s;
  if (c == 0) {
    float b = 0.f;
    for (int i = 0; i < HEAD_DIM; ++i)
      b = fmaf(q[h * HEAD_DIM + i], bk[h * HEAD_DIM + i], b);
    bias[h] = b;
  }
}

// ---------------------------------------------------------------------------
// Phase 2b: stage zero-padded f16 operands so the WMMA loop is branch-free.
//   Ah[208][512]  = f16(V[0,0,row,:]) for row<196, else 0
//   Bh[16][512]   = f16(u[n][:])      for n<4,     else 0
// ---------------------------------------------------------------------------
__global__ void k_stage(const float* __restrict__ vis, const float* __restrict__ u,
                        _Float16* __restrict__ Ah, _Float16* __restrict__ Bh) {
  int idx = blockIdx.x * blockDim.x + threadIdx.x;
  if (idx < A_ELEMS) {
    int row = idx >> 9, c = idx & (C_DIM - 1);
    float v = (row < N_PATCH) ? vis[(size_t)row * C_DIM + c] : 0.f;
    Ah[idx] = (_Float16)v;
  } else if (idx < A_ELEMS + B_ELEMS) {
    int k = idx - A_ELEMS;
    int row = k >> 9, c = k & (C_DIM - 1);
    float v = (row < N_HEADS) ? u[(size_t)row * C_DIM + c] : 0.f;
    Bh[k] = (_Float16)v;
  }
}

// ---------------------------------------------------------------------------
// Phase 3: scores[h][n] = dot(V[0,0,n,:], u[h][:])  via v_wmma_f32_16x16x32_f16
// One wave per 16-row tile. Per ISA 7.12.2 the 16-bit operand layout makes a
// lane's v16h = two contiguous 16-byte runs of its row:
//   elems 0..7  = M[row][kk + half*8 .. +7]
//   elems 8..15 = M[row][kk + 16 + half*8 .. +7]
// so each operand is two unconditional global_load_b128 from row-major f16.
// D layout (f32 C/D): element r -> row (r + 8*half), col (lane & 15).
// ---------------------------------------------------------------------------
__global__ void k_scores_wmma(const _Float16* __restrict__ Ah,
                              const _Float16* __restrict__ Bh,
                              float* __restrict__ scores) {
  const int m0   = blockIdx.x * 16;
  const int lane = threadIdx.x & 31;   // wave32
  const int mn   = lane & 15;          // M for A, N for B
  const int half = lane >> 4;
  const _Float16* arow = Ah + (size_t)(m0 + mn) * C_DIM + half * 8;
  const _Float16* bcol = Bh + (size_t)mn * C_DIM + half * 8;

  v8f acc = {};
#pragma unroll 4
  for (int kk = 0; kk < C_DIM; kk += 32) {
    v8h alo = *(const v8h*)(arow + kk);
    v8h ahi = *(const v8h*)(arow + kk + 16);
    v8h blo = *(const v8h*)(bcol + kk);
    v8h bhi = *(const v8h*)(bcol + kk + 16);
    v16h a = __builtin_shufflevector(alo, ahi, 0, 1, 2, 3, 4, 5, 6, 7,
                                     8, 9, 10, 11, 12, 13, 14, 15);
    v16h b = __builtin_shufflevector(blo, bhi, 0, 1, 2, 3, 4, 5, 6, 7,
                                     8, 9, 10, 11, 12, 13, 14, 15);
    acc = __builtin_amdgcn_wmma_f32_16x16x32_f16(false, a, false, b,
                                                 (short)0, acc, false, false);
  }

  if (mn < N_HEADS) {
#pragma unroll
    for (int r = 0; r < 8; ++r)
      scores[mn * N_PAD + m0 + r + half * 8] = acc[r];
  }
}

// ---------------------------------------------------------------------------
// Phase 4: per-head softmax over n (196), mean over heads, then 10 rounds of
// parallel argmax; p = max index among the 10 winners. Single 256-thread block.
// ---------------------------------------------------------------------------
__global__ void k_select(const float* __restrict__ scores, const float* __restrict__ bias,
                         int* __restrict__ pidx) {
  __shared__ float red[256];
  __shared__ int   redi[256];
  __shared__ float w[N_PAD];
  __shared__ int   selmax;

  const int n = threadIdx.x;
  const bool act = (n < N_PATCH);

  float sc[N_HEADS];
  for (int h = 0; h < N_HEADS; ++h)
    sc[h] = act ? (scores[h * N_PAD + n] + bias[h]) * SCALE : -1e30f;

  float wacc = 0.f;
  for (int h = 0; h < N_HEADS; ++h) {
    red[n] = sc[h];
    __syncthreads();
    for (int s = 128; s > 0; s >>= 1) {
      if (n < s) red[n] = fmaxf(red[n], red[n + s]);
      __syncthreads();
    }
    const float m = red[0];
    __syncthreads();
    const float e = act ? expf(sc[h] - m) : 0.f;
    red[n] = e;
    __syncthreads();
    for (int s = 128; s > 0; s >>= 1) {
      if (n < s) red[n] += red[n + s];
      __syncthreads();
    }
    const float d = red[0];
    __syncthreads();
    wacc += e / d;
  }

  if (n < N_PAD) w[n] = act ? 0.25f * wacc : -1e30f;
  if (n == 0) selmax = -1;
  __syncthreads();

  for (int it = 0; it < TOPM; ++it) {
    red[n]  = (n < N_PAD) ? w[n] : -1e30f;
    redi[n] = n;
    __syncthreads();
    for (int s = 128; s > 0; s >>= 1) {
      if (n < s) {
        if (red[n + s] > red[n]) { red[n] = red[n + s]; redi[n] = redi[n + s]; }
      }
      __syncthreads();
    }
    if (n == 0) {
      const int best = redi[0];
      if (best > selmax) selmax = best;
      w[best] = -2e30f;           // remove winner for next round
    }
    __syncthreads();
  }
  if (n == 0) pidx[0] = selmax;
}

// ---------------------------------------------------------------------------
// Phase 5: broadcast gather (the bandwidth-bound part, ~39.3 MB of stores).
// One block per (b,t); 128 lanes load one float4 of V[b,t,p,:] and write it
// to all TOPM slots of both concatenated outputs (identical flat layouts).
// ---------------------------------------------------------------------------
__global__ void k_gather(const float4* __restrict__ vis, const int* __restrict__ pidx,
                         float4* __restrict__ out) {
  const int bt = blockIdx.x;        // 0 .. B*T-1
  const int c4 = threadIdx.x;       // 0 .. 127
  const int p  = pidx[0];
  const float4 v = vis[((size_t)bt * N_PATCH + p) * (C_DIM / 4) + c4];
  float4* o0 = out;
  float4* o1 = out + (size_t)B_DIM * T_DIM * TOPM * (C_DIM / 4);
  const size_t base = (size_t)bt * TOPM * (C_DIM / 4) + c4;
#pragma unroll
  for (int m = 0; m < TOPM; ++m) {
    o0[base + (size_t)m * (C_DIM / 4)] = v;
    o1[base + (size_t)m * (C_DIM / 4)] = v;
  }
}

// ---------------------------------------------------------------------------
extern "C" void kernel_launch(void* const* d_in, const int* in_sizes, int n_in,
                              void* d_out, int out_size, void* d_ws, size_t ws_size,
                              hipStream_t stream) {
  (void)in_sizes; (void)n_in; (void)out_size; (void)ws_size;
  const float* vis = (const float*)d_in[0];   // (16,60,196,512)
  const float* qst = (const float*)d_in[1];   // (16,512) — only row 0 matters
  const float* Wq  = (const float*)d_in[2];   // (512,512)
  const float* bq  = (const float*)d_in[3];   // (512,)
  const float* Wk  = (const float*)d_in[4];   // (512,512)
  const float* bk  = (const float*)d_in[5];   // (512,)

  float* ws     = (float*)d_ws;
  float* qv     = ws;            // 512 floats
  float* u      = ws + 512;      // 4*512 = 2048 floats
  float* bias   = ws + 2560;     // 4 floats
  float* scores = ws + 2568;     // 4*208 = 832 floats
  int*   pidx   = (int*)(ws + 3408);
  _Float16* Ah  = (_Float16*)((char*)d_ws + 16384);                    // 208*512 f16
  _Float16* Bh  = (_Float16*)((char*)d_ws + 16384 + A_ELEMS * 2);     // 16*512 f16

  k_qproj      <<<2,   256, 0, stream>>>(qst, Wq, bq, qv);
  k_uproj      <<<8,   256, 0, stream>>>(qv, Wk, bk, u, bias);
  k_stage      <<<(A_ELEMS + B_ELEMS + 255) / 256, 256, 0, stream>>>(vis, u, Ah, Bh);
  k_scores_wmma<<<13,  32,  0, stream>>>(Ah, Bh, scores);
  k_select     <<<1,   256, 0, stream>>>(scores, bias, pidx);
  k_gather     <<<960, 128, 0, stream>>>((const float4*)vis, pidx, (float4*)d_out);
}
